// TimingMPNN_78365973283355
// MI455X (gfx1250) — compile-verified, matrix operand order checked by reading, not run on hardware
//
#include <hip/hip_runtime.h>
#include <hip/hip_bf16.h>
#include <stdint.h>

// ---------------------------------------------------------------------------
// MPNN forward for MI455X (gfx1250): f16 WMMA (f32 accum) for all GEMMs,
// hardware f32 atomics (global_atomic_add_f32 @ DEV scope) for segment_sum,
// fused gather->MLP->scatter edge kernel.
// ---------------------------------------------------------------------------

#define NN 40000
#define NE 400000

typedef __attribute__((ext_vector_type(16))) _Float16 v16h;
typedef __attribute__((ext_vector_type(8)))  _Float16 v8h;
typedef __attribute__((ext_vector_type(4)))  _Float16 v4h;
typedef __attribute__((ext_vector_type(8)))  float    v8f;
typedef __attribute__((ext_vector_type(4)))  float    v4f;

static __device__ __forceinline__ v8f wmma_f16(v16h a, v16h b, v8f c) {
  return __builtin_amdgcn_wmma_f32_16x16x32_f16(false, a, false, b, (short)0, c,
                                                false, false);
}

// Guaranteed single-instruction f32 atomic add (no-return form -> STOREcnt),
// device scope so the RMW executes at the L2 atomic units.  No "memory"
// clobber: the only consumer is a later kernel (stream-ordered; S_ENDPGM
// drains STOREcnt), and `volatile` already pins emission + mutual order.
static __device__ __forceinline__ void atomic_add_f32(float* p, float v) {
  asm volatile("global_atomic_add_f32 %0, %1, off scope:SCOPE_DEV"
               :
               : "v"(p), "v"(v));
}

// A-matrix 16x32 f16 chunk: lane holds row m = lane&15, half = lane>>4.
// K pairs: j=0..3 -> K = 2j + 8*half ; j=4..7 -> K = 2j + 8 + 8*half.
// => elements [0..7] = row[k0 + 8h .. k0+8h+7], [8..15] = row[k0+16+8h ..].
static __device__ __forceinline__ v16h load_a16(const _Float16* row, int k0, int half) {
  v8h lo = *reinterpret_cast<const v8h*>(row + k0 + 8 * half);
  v8h hi = *reinterpret_cast<const v8h*>(row + k0 + 16 + 8 * half);
  v16h r;
#pragma unroll
  for (int i = 0; i < 8; ++i) { r[i] = lo[i]; r[i + 8] = hi[i]; }
  return r;
}

// Same A layout but source row is f32 (converted on the fly).
static __device__ __forceinline__ v16h load_a32(const float* row, int k0, int half) {
  v4f a0 = *reinterpret_cast<const v4f*>(row + k0 + 8 * half);
  v4f a1 = *reinterpret_cast<const v4f*>(row + k0 + 8 * half + 4);
  v4f b0 = *reinterpret_cast<const v4f*>(row + k0 + 16 + 8 * half);
  v4f b1 = *reinterpret_cast<const v4f*>(row + k0 + 16 + 8 * half + 4);
  v16h r;
#pragma unroll
  for (int i = 0; i < 4; ++i) {
    r[i]      = (_Float16)a0[i];
    r[i + 4]  = (_Float16)a1[i];
    r[i + 8]  = (_Float16)b0[i];
    r[i + 12] = (_Float16)b1[i];
  }
  return r;
}

// Pre-packed B: [(t*nch + c)*32 + lane] * 8 dwords, contiguous 32B per lane.
static __device__ __forceinline__ v16h load_b(const uint32_t* Bp, int t, int c,
                                              int nch, int lane) {
  return *reinterpret_cast<const v16h*>(Bp + (size_t)((t * nch + c) * 32 + lane) * 8);
}

static __device__ __forceinline__ uint32_t pack2h(_Float16 a, _Float16 b) {
  unsigned short x = __builtin_bit_cast(unsigned short, a);
  unsigned short y = __builtin_bit_cast(unsigned short, b);
  return (uint32_t)x | ((uint32_t)y << 16);
}

// ---------------------------------------------------------------------------
// Weight packing: W is [K x N] row-major f32 -> WMMA-B f16 layout, zero-padded.
// ---------------------------------------------------------------------------
__global__ void pack_b_k(const float* __restrict__ W, int K, int N, int nch,
                         int nt, uint32_t* __restrict__ out) {
  int idx = blockIdx.x * blockDim.x + threadIdx.x;
  int total = nt * nch * 32;
  if (idx >= total) return;
  int lane = idx & 31;
  int c = (idx >> 5) % nch;
  int t = (idx >> 5) / nch;
  int half = lane >> 4;
  int n = t * 16 + (lane & 15);
  uint32_t* o = out + (size_t)idx * 8;
#pragma unroll
  for (int j = 0; j < 8; ++j) {
    int e0 = 2 * j, e1 = 2 * j + 1;
    int k0 = 32 * c + 8 * half + (e0 < 8 ? e0 : e0 + 8);
    int k1 = 32 * c + 8 * half + (e1 < 8 ? e1 : e1 + 8);
    float f0 = (k0 < K && n < N) ? W[(size_t)k0 * N + n] : 0.f;
    float f1 = (k1 < K && n < N) ? W[(size_t)k1 * N + n] : 0.f;
    o[j] = pack2h((_Float16)f0, (_Float16)f1);
  }
}

// f32 [rows x K] -> f16 [rows x 32] zero-padded (encoder inputs, K=14 / K=12).
__global__ void cvt_pad_k(const float* __restrict__ X, int rows, int K,
                          _Float16* __restrict__ out) {
  int idx = blockIdx.x * blockDim.x + threadIdx.x;
  if (idx >= rows * 32) return;
  int r = idx >> 5, k = idx & 31;
  out[idx] = (k < K) ? (_Float16)X[(size_t)r * K + k] : (_Float16)0.f;
}

__global__ void zero_k(float* p, int n4) {
  int i = blockIdx.x * blockDim.x + threadIdx.x;
  if (i < n4) reinterpret_cast<v4f*>(p)[i] = (v4f){0.f, 0.f, 0.f, 0.f};
}

// ---------------------------------------------------------------------------
// Generic GEMM: C = act(A16 @ W + bias).  One wave per 16-row tile.
// A kept fully in registers (=> in-place A/out aliasing on same rows is safe).
// ---------------------------------------------------------------------------
template <int NCH, int NT, bool RELU, bool OUTF, bool OUTH>
__global__ __launch_bounds__(128) void wmma_gemm(const _Float16* A,
                                                 const uint32_t* __restrict__ Bp,
                                                 const float* __restrict__ bias,
                                                 float* Cf, _Float16* Ch,
                                                 int ntiles) {
  const int lane = threadIdx.x & 31;
  const int wid = threadIdx.x >> 5;
  const int tile = blockIdx.x * 4 + wid;
  if (tile >= ntiles) return;  // wave-uniform
  const int m = lane & 15, half = lane >> 4;
  const int K = NCH * 32, N = NT * 16;
  const _Float16* row = A + (size_t)(tile * 16 + m) * K;
  v16h a[NCH];
#pragma unroll
  for (int c = 0; c < NCH; ++c) a[c] = load_a16(row, 32 * c, half);
#pragma unroll 2
  for (int t = 0; t < NT; ++t) {
    v8f acc = {};
#pragma unroll
    for (int c = 0; c < NCH; ++c) acc = wmma_f16(a[c], load_b(Bp, t, c, NCH, lane), acc);
    const float bv = bias[t * 16 + m];
#pragma unroll
    for (int r = 0; r < 8; ++r) {
      float v = acc[r] + bv;
      if (RELU) v = fmaxf(v, 0.f);
      const size_t o = (size_t)(tile * 16 + r + 8 * half) * N + (t * 16 + m);
      if constexpr (OUTF) Cf[o] = v;
      if constexpr (OUTH) Ch[o] = (_Float16)v;
    }
  }
}

// ---------------------------------------------------------------------------
// Edge kernel: m = MLP([h[src]|h[dst]|e]) then atomic scatter-add into agg[dst].
// One wave per 16 edges; 4 waves per block.
// ---------------------------------------------------------------------------
__global__ __launch_bounds__(128) void edge_msg_k(
    const _Float16* __restrict__ h16, const _Float16* __restrict__ e16,
    const int* __restrict__ src, const int* __restrict__ dst,
    const uint32_t* __restrict__ B1, const float* __restrict__ b1,
    const uint32_t* __restrict__ B2, const float* __restrict__ b2,
    float* __restrict__ agg, int ntiles) {
  __shared__ __align__(16) _Float16 Tl[4][16][256];
  const int lane = threadIdx.x & 31;
  const int wid = threadIdx.x >> 5;
  const int tile = blockIdx.x * 4 + wid;
  if (tile >= ntiles) return;
  const int m = lane & 15, half = lane >> 4;
  const int e0 = tile * 16;
  const int si = src[e0 + m];
  const int di = dst[e0 + m];
  const _Float16* pS = h16 + (size_t)si * 128;
  const _Float16* pD = h16 + (size_t)di * 128;
  const _Float16* pE = e16 + (size_t)(e0 + m) * 128;

  v16h a[12];
#pragma unroll
  for (int c = 0; c < 4; ++c) {
    a[c]     = load_a16(pS, 32 * c, half);
    a[4 + c] = load_a16(pD, 32 * c, half);
    a[8 + c] = load_a16(pE, 32 * c, half);
  }
  // GEMM1: [16x384] @ [384x256] -> ReLU -> LDS (f16, row-major 16x256)
#pragma unroll 2
  for (int t = 0; t < 16; ++t) {
    v8f acc = {};
#pragma unroll
    for (int c = 0; c < 12; ++c) acc = wmma_f16(a[c], load_b(B1, t, c, 12, lane), acc);
    const float bv = b1[t * 16 + m];
#pragma unroll
    for (int r = 0; r < 8; ++r) {
      float v = fmaxf(acc[r] + bv, 0.f);
      Tl[wid][r + 8 * half][t * 16 + m] = (_Float16)v;
    }
  }
  // GEMM2: [16x256] @ [256x128] + b2, scatter rows to agg[dst]
  v16h a2[8];
#pragma unroll
  for (int c = 0; c < 8; ++c) a2[c] = load_a16(&Tl[wid][m][0], 32 * c, half);

  int4 d0 = *reinterpret_cast<const int4*>(dst + e0 + 8 * half);
  int4 d1 = *reinterpret_cast<const int4*>(dst + e0 + 8 * half + 4);
  int drow[8] = {d0.x, d0.y, d0.z, d0.w, d1.x, d1.y, d1.z, d1.w};

#pragma unroll
  for (int t = 0; t < 8; ++t) {
    v8f acc = {};
#pragma unroll
    for (int c = 0; c < 8; ++c) acc = wmma_f16(a2[c], load_b(B2, t, c, 8, lane), acc);
    const float bv = b2[t * 16 + m];
#pragma unroll
    for (int r = 0; r < 8; ++r) {
      atomic_add_f32(&agg[(size_t)drow[r] * 128 + (t * 16 + m)], acc[r] + bv);
    }
  }
}

// ---------------------------------------------------------------------------
// Node kernel: h_upd = MLP([h|agg]); hres = h + h_upd. One wave per 16 nodes.
// ---------------------------------------------------------------------------
__global__ __launch_bounds__(128) void node_upd_k(
    const _Float16* __restrict__ h16, const float* __restrict__ hF,
    const float* __restrict__ agg, const uint32_t* __restrict__ B1,
    const float* __restrict__ b1, const uint32_t* __restrict__ B2,
    const float* __restrict__ b2, float* __restrict__ hres, int ntiles) {
  __shared__ __align__(16) _Float16 Tl[4][16][256];
  const int lane = threadIdx.x & 31;
  const int wid = threadIdx.x >> 5;
  const int tile = blockIdx.x * 4 + wid;
  if (tile >= ntiles) return;
  const int m = lane & 15, half = lane >> 4;
  const int n0 = tile * 16;
  const _Float16* pH = h16 + (size_t)(n0 + m) * 128;
  const float* pA = agg + (size_t)(n0 + m) * 128;

  v16h a[8];
#pragma unroll
  for (int c = 0; c < 4; ++c) {
    a[c]     = load_a16(pH, 32 * c, half);
    a[4 + c] = load_a32(pA, 32 * c, half);
  }
#pragma unroll 2
  for (int t = 0; t < 16; ++t) {
    v8f acc = {};
#pragma unroll
    for (int c = 0; c < 8; ++c) acc = wmma_f16(a[c], load_b(B1, t, c, 8, lane), acc);
    const float bv = b1[t * 16 + m];
#pragma unroll
    for (int r = 0; r < 8; ++r) {
      float v = fmaxf(acc[r] + bv, 0.f);
      Tl[wid][r + 8 * half][t * 16 + m] = (_Float16)v;
    }
  }
  v16h a2[8];
#pragma unroll
  for (int c = 0; c < 8; ++c) a2[c] = load_a16(&Tl[wid][m][0], 32 * c, half);

#pragma unroll
  for (int t = 0; t < 8; ++t) {
    v8f acc = {};
#pragma unroll
    for (int c = 0; c < 8; ++c) acc = wmma_f16(a2[c], load_b(B2, t, c, 8, lane), acc);
    const float bv = b2[t * 16 + m];
#pragma unroll
    for (int r = 0; r < 8; ++r) {
      const size_t o = (size_t)(n0 + r + 8 * half) * 128 + (t * 16 + m);
      hres[o] = hF[o] + (acc[r] + bv);
    }
  }
}

// ---------------------------------------------------------------------------
// LayerNorm: one wave per row of 128; writes f32 h and f16 h16.
// ---------------------------------------------------------------------------
__global__ __launch_bounds__(256) void ln_k(const float* __restrict__ hres,
                                            const float* __restrict__ g,
                                            const float* __restrict__ b,
                                            float* __restrict__ hF,
                                            _Float16* __restrict__ h16, int nrows) {
  const int lane = threadIdx.x & 31;
  const int wid = threadIdx.x >> 5;
  const int row = blockIdx.x * 8 + wid;
  if (row >= nrows) return;
  v4f x = *reinterpret_cast<const v4f*>(hres + (size_t)row * 128 + lane * 4);
  float s = x[0] + x[1] + x[2] + x[3];
#pragma unroll
  for (int msk = 16; msk >= 1; msk >>= 1) s += __shfl_xor(s, msk, 32);
  const float mu = s * (1.f / 128.f);
  v4f d;
#pragma unroll
  for (int i = 0; i < 4; ++i) d[i] = x[i] - mu;
  float q = d[0] * d[0] + d[1] * d[1] + d[2] * d[2] + d[3] * d[3];
#pragma unroll
  for (int msk = 16; msk >= 1; msk >>= 1) q += __shfl_xor(q, msk, 32);
  const float rs = rsqrtf(q * (1.f / 128.f) + 1e-5f);
  v4f gg = *reinterpret_cast<const v4f*>(g + lane * 4);
  v4f bb = *reinterpret_cast<const v4f*>(b + lane * 4);
  v4f y;
  v4h yh;
#pragma unroll
  for (int i = 0; i < 4; ++i) {
    y[i] = d[i] * rs * gg[i] + bb[i];
    yh[i] = (_Float16)y[i];
  }
  *reinterpret_cast<v4f*>(hF + (size_t)row * 128 + lane * 4) = y;
  *reinterpret_cast<v4h*>(h16 + (size_t)row * 128 + lane * 4) = yh;
}

// Final head layer: out = t2 @ W2(64x1) + b2
__global__ void head_final_k(const float* __restrict__ t2,
                             const float* __restrict__ w2,
                             const float* __restrict__ b2,
                             float* __restrict__ out, int n) {
  int i = blockIdx.x * blockDim.x + threadIdx.x;
  if (i >= n) return;
  float s = 0.f;
#pragma unroll
  for (int k = 0; k < 64; ++k) s += t2[(size_t)i * 64 + k] * w2[k];
  out[i] = s + b2[0];
}

// ---------------------------------------------------------------------------
// Host side
// ---------------------------------------------------------------------------
extern "C" void kernel_launch(void* const* d_in, const int* in_sizes, int n_in,
                              void* d_out, int out_size, void* d_ws, size_t ws_size,
                              hipStream_t stream) {
  (void)out_size; (void)ws_size;
  // --- resolve input ordering by unique element counts ---
  int ix = -1, iei = -1, iea = -1, ine = -1, iee = -1, irh = -1;
  int lbase[6]; int nl = 0;
  for (int i = 0; i < n_in; ++i) {
    switch (in_sizes[i]) {
      case NN * 14:   ix = i;  break;
      case 2 * NE:    iei = i; break;
      case NE * 12:   iea = i; break;
      case 14 * 128:  ine = i; break;
      case 12 * 128:  iee = i; break;
      case 128 * 64:  irh = i; break;
      case 384 * 256: if (nl < 6) lbase[nl++] = i; break;
      default: break;
    }
  }
  auto F = [&](int i) { return (const float*)d_in[i]; };
  const float* x = F(ix);
  const int* src = (const int*)d_in[iei];
  const int* dst = src + NE;
  const float* eattr = F(iea);
  const float *neW1 = F(ine), *neb1 = F(ine + 1), *neW2 = F(ine + 2), *neb2 = F(ine + 3);
  const float *eeW1 = F(iee), *eeb1 = F(iee + 1), *eeW2 = F(iee + 2), *eeb2 = F(iee + 3);
  const float *rW1 = F(irh), *rb1 = F(irh + 1), *rW2 = F(irh + 2), *rb2 = F(irh + 3);
  struct Lp { const float *eW1, *eb1, *eW2, *eb2, *nW1, *nb1, *nW2, *nb2, *g, *b; } L[6];
  for (int l = 0; l < 6; ++l) {
    int b0 = lbase[l];
    if (in_sizes[b0 + 1] == 256 * 128) {  // jax-sorted key order
      L[l].eW1 = F(b0);     L[l].eW2 = F(b0 + 1);
      L[l].eb1 = F(b0 + 2); L[l].eb2 = F(b0 + 3);
      L[l].b = F(b0 + 4);   L[l].g = F(b0 + 5);
      L[l].nW1 = F(b0 + 6); L[l].nW2 = F(b0 + 7);
      L[l].nb1 = F(b0 + 8); L[l].nb2 = F(b0 + 9);
    } else {  // insertion order
      L[l].eW1 = F(b0);     L[l].eb1 = F(b0 + 1);
      L[l].eW2 = F(b0 + 2); L[l].eb2 = F(b0 + 3);
      L[l].nW1 = F(b0 + 4); L[l].nb1 = F(b0 + 5);
      L[l].nW2 = F(b0 + 6); L[l].nb2 = F(b0 + 7);
      L[l].g = F(b0 + 8);   L[l].b = F(b0 + 9);
    }
  }

  // --- workspace bump allocator ---
  size_t off = 0;
  auto alloc = [&](size_t bytes) -> void* {
    void* p = (void*)((char*)d_ws + off);
    off += (bytes + 255) & ~(size_t)255;
    return p;
  };
  float*     hF   = (float*)alloc((size_t)NN * 128 * 4);
  _Float16*  h16  = (_Float16*)alloc((size_t)NN * 128 * 2);
  float*     hres = (float*)alloc((size_t)NN * 128 * 4);
  float*     agg  = (float*)alloc((size_t)NN * 128 * 4);
  _Float16*  e16  = (_Float16*)alloc((size_t)NE * 128 * 2);
  _Float16*  x16  = (_Float16*)alloc((size_t)NN * 32 * 2);
  _Float16*  ea16 = (_Float16*)alloc((size_t)NE * 32 * 2);
  float*     t2   = (float*)alloc((size_t)NN * 64 * 4);

  auto pack = [&](const float* W, int K, int N, int nch, int nt) -> const uint32_t* {
    uint32_t* o = (uint32_t*)alloc((size_t)nt * nch * 256 * 4);
    int total = nt * nch * 32;
    pack_b_k<<<(total + 255) / 256, 256, 0, stream>>>(W, K, N, nch, nt, o);
    return o;
  };
  const uint32_t* pNe1 = pack(neW1, 14, 128, 1, 8);
  const uint32_t* pNe2 = pack(neW2, 128, 128, 4, 8);
  const uint32_t* pEe1 = pack(eeW1, 12, 128, 1, 8);
  const uint32_t* pEe2 = pack(eeW2, 128, 128, 4, 8);
  const uint32_t *pE1[6], *pE2[6], *pN1[6], *pN2[6];
  for (int l = 0; l < 6; ++l) {
    pE1[l] = pack(L[l].eW1, 384, 256, 12, 16);
    pE2[l] = pack(L[l].eW2, 256, 128, 8, 8);
    pN1[l] = pack(L[l].nW1, 256, 256, 8, 16);
    pN2[l] = pack(L[l].nW2, 256, 128, 8, 8);
  }
  const uint32_t* pRh = pack(rW1, 128, 64, 4, 4);

  // --- encoders ---
  cvt_pad_k<<<(NN * 32 + 255) / 256, 256, 0, stream>>>(x, NN, 14, x16);
  cvt_pad_k<<<(NE * 32 + 255) / 256, 256, 0, stream>>>(eattr, NE, 12, ea16);

  const int NT_N = NN / 16;   // 2500 node tiles
  const int NT_E = NE / 16;   // 25000 edge tiles
  wmma_gemm<1, 8, true,  false, true ><<<NT_N / 4, 128, 0, stream>>>(x16, pNe1, neb1, nullptr, h16, NT_N);
  wmma_gemm<4, 8, false, true,  true ><<<NT_N / 4, 128, 0, stream>>>(h16, pNe2, neb2, hF, h16, NT_N);
  wmma_gemm<1, 8, true,  false, true ><<<NT_E / 4, 128, 0, stream>>>(ea16, pEe1, eeb1, nullptr, e16, NT_E);
  wmma_gemm<4, 8, false, false, true ><<<NT_E / 4, 128, 0, stream>>>(e16, pEe2, eeb2, nullptr, e16, NT_E);

  // --- message passing layers ---
  for (int l = 0; l < 6; ++l) {
    zero_k<<<(NN * 128 / 4 + 255) / 256, 256, 0, stream>>>(agg, NN * 128 / 4);
    edge_msg_k<<<NT_E / 4, 128, 0, stream>>>(h16, e16, src, dst, pE1[l], L[l].eb1,
                                             pE2[l], L[l].eb2, agg, NT_E);
    node_upd_k<<<NT_N / 4, 128, 0, stream>>>(h16, hF, agg, pN1[l], L[l].nb1,
                                             pN2[l], L[l].nb2, hres, NT_N);
    ln_k<<<NN / 8, 256, 0, stream>>>(hres, L[l].g, L[l].b, hF, h16, NN);
  }

  // --- regression head ---
  wmma_gemm<4, 4, true, true, false><<<NT_N / 4, 128, 0, stream>>>(h16, pRh, rb1, t2, nullptr, NT_N);
  head_final_k<<<(NN + 255) / 256, 256, 0, stream>>>(t2, rW2, rb2, (float*)d_out, NN);
}
